// StreamingTransformerLayer_2989297238372
// MI455X (gfx1250) — matrix operand order repaired and checked by reading, not compiled, not run on hardware
//
#include <hip/hip_runtime.h>

// Problem constants (match reference)
constexpr int kB   = 4;
constexpr int kT   = 1024;
constexpr int kD   = 1024;
constexpr int kH   = 16;
constexpr int kFF  = 4096;
constexpr int kCAP = 2048;
constexpr int kDH  = 64;      // kD / kH
constexpr int kCTX = 2048;
constexpr int kBT  = kB * kT;

typedef __attribute__((ext_vector_type(16))) _Float16 v16h;
typedef __attribute__((ext_vector_type(8)))  _Float16 v8h;
typedef __attribute__((ext_vector_type(8)))  float    v8f;

__device__ __forceinline__ v8f wmma_f16(v16h a, v16h b, v8f c) {
  // D = A(16x32 f16) * B(32x16 f16) + C(16x16 f32)
  return __builtin_amdgcn_wmma_f32_16x16x32_f16(false, a, false, b, (short)0, c,
                                                false, false);
}

// A-matrix fragment (16x32 f16). Source row-major [*, lda].
// Lane l: row m0 + (l&15); halves j<8 at k = k0 + hi*8 + j, j>=8 at k0+16+hi*8+(j-8).
__device__ __forceinline__ v16h load_a_frag(const _Float16* A, int lda, int m0,
                                            int k0, int lane) {
  int m  = m0 + (lane & 15);
  int hi = (lane >> 4) & 1;
  const _Float16* p = A + (size_t)m * lda + k0 + hi * 8;
  v8h lo = *(const v8h*)(p);
  v8h hh = *(const v8h*)(p + 16);
  v16h a;
#pragma unroll
  for (int i = 0; i < 8; ++i) { a[i] = lo[i]; a[i + 8] = hh[i]; }
  return a;
}

// B-matrix fragment (32x16 f16): B[k][n] = W[n0+n][k0+k], W row-major [N, ldw].
// Lane l: n = l&15, khi = l>>4; halves j<8 at k = khi*8 + j, j>=8 at 16+khi*8+(j-8).
__device__ __forceinline__ v16h load_b_frag(const _Float16* W, int ldw, int n0,
                                            int k0, int lane) {
  int n   = n0 + (lane & 15);
  int khi = (lane >> 4) & 1;
  const _Float16* p = W + (size_t)n * ldw + k0 + khi * 8;
  v8h lo = *(const v8h*)(p);
  v8h hh = *(const v8h*)(p + 16);
  v16h b;
#pragma unroll
  for (int i = 0; i < 8; ++i) { b[i] = lo[i]; b[i + 8] = hh[i]; }
  return b;
}

// ---------------------------------------------------------------- elementwise
__global__ void f32_to_f16_kernel(const float* __restrict__ in,
                                  _Float16* __restrict__ out, int n) {
  int i = blockIdx.x * 256 + threadIdx.x;
  if (i < n) out[i] = (_Float16)in[i];
}

// Convert KV cache (f32, [2,B,H,CAP,DH]) into f16 K ([B,H,CAP,DH]) and
// transposed V ([B,H,DH,CAP]).
__global__ void cache_cvt_kernel(const float* __restrict__ cache,
                                 _Float16* __restrict__ ck16,
                                 _Float16* __restrict__ cvT, int total) {
  int i = blockIdx.x * 256 + threadIdx.x;
  if (i >= total) return;
  ck16[i] = (_Float16)cache[i];
  int d  = i & (kDH - 1);
  int s  = (i >> 6) & (kCAP - 1);
  int bh = i >> 17;  // DH*CAP = 2^17
  cvT[((size_t)bh * kDH + d) * kCAP + s] = (_Float16)cache[(size_t)total + i];
}

// positions[b][slot] per reference ring-buffer logic
__global__ void pos_kernel(const int* __restrict__ eoff, int* __restrict__ pos) {
  int i = blockIdx.x * 256 + threadIdx.x;  // over B*CAP
  int b = i / kCAP, s = i % kCAP;
  int e = eoff[b];
  int last = e + kT - 1;
  int end_index = last % kCAP;
  int delta = s - end_index;
  int p = (delta <= 0) ? (last + delta) : (last + delta - kCAP);
  if (s >= e + kT) p = -1;
  pos[i] = p;
}

// ----------------------------------------------------------------- layernorm
__global__ __launch_bounds__(256)
void ln_kernel(const float* __restrict__ x, const float* __restrict__ g,
               const float* __restrict__ be, _Float16* __restrict__ out, int Dn) {
  __shared__ float sred[2][8];
  int row = blockIdx.x;
  const float* xr = x + (size_t)row * Dn;
  float s = 0.f, s2 = 0.f;
  for (int i = threadIdx.x; i < Dn; i += 256) {
    float v = xr[i]; s += v; s2 += v * v;
  }
#pragma unroll
  for (int msk = 16; msk >= 1; msk >>= 1) {
    s  += __shfl_xor(s,  msk, 32);
    s2 += __shfl_xor(s2, msk, 32);
  }
  int lane = threadIdx.x & 31, w = threadIdx.x >> 5;
  if (lane == 0) { sred[0][w] = s; sred[1][w] = s2; }
  __syncthreads();
  float ts = 0.f, ts2 = 0.f;
#pragma unroll
  for (int i = 0; i < 8; ++i) { ts += sred[0][i]; ts2 += sred[1][i]; }
  float mean = ts / Dn;
  float var  = ts2 / Dn - mean * mean;
  float rstd = rsqrtf(var + 1e-5f);
  for (int i = threadIdx.x; i < Dn; i += 256)
    out[(size_t)row * Dn + i] = (_Float16)((xr[i] - mean) * rstd * g[i] + be[i]);
}

// --------------------------------------------------------------------- GEMM
// C[m,n] = sum_k A[m,k] * W[n,k]; per-wave 32x64 tile (2 A frags reuse each
// B frag -> 8 WMMAs per 4 B loads), block = 4 waves = 32x256 tile.
// EPI: 0 = f16 store, 1 = exact GELU -> f16, 2 = f32 residual add -> f32
template <int EPI>
__global__ __launch_bounds__(128)
void gemm_kernel(const _Float16* __restrict__ A, const _Float16* __restrict__ W,
                 int M, int N, int K, _Float16* __restrict__ outH,
                 float* __restrict__ outF, const float* __restrict__ resid) {
  int lane = threadIdx.x & 31;
  int wave = threadIdx.x >> 5;
  int m0 = blockIdx.y * 32;
  int n0 = blockIdx.x * 256 + wave * 64;
  v8f acc[2][4] = {};
  for (int k0 = 0; k0 < K; k0 += 32) {
    v16h a0 = load_a_frag(A, K, m0,      k0, lane);
    v16h a1 = load_a_frag(A, K, m0 + 16, k0, lane);
    if (k0 + 32 < K) {
      // pull next K-panel of B toward the WGP while the WMMAs run
      __builtin_prefetch(W + (size_t)(n0 + (lane & 15)) * K + k0 + 32 +
                             ((lane >> 4) & 1) * 16, 0, 1);
    }
#pragma unroll
    for (int j = 0; j < 4; ++j) {
      v16h b = load_b_frag(W, K, n0 + j * 16, k0, lane);
      acc[0][j] = wmma_f16(a0, b, acc[0][j]);
      acc[1][j] = wmma_f16(a1, b, acc[1][j]);
    }
  }
  int n = lane & 15, hi = lane >> 4;
#pragma unroll
  for (int i = 0; i < 2; ++i) {
#pragma unroll
    for (int j = 0; j < 4; ++j) {
#pragma unroll
      for (int r = 0; r < 8; ++r) {
        size_t idx = (size_t)(m0 + i * 16 + r + 8 * hi) * N + (n0 + j * 16 + n);
        float v = acc[i][j][r];
        if (EPI == 1) {
          v = 0.5f * v * (1.0f + erff(v * 0.70710678118f));
          outH[idx] = (_Float16)v;
        } else if (EPI == 2) {
          outF[idx] = resid[idx] + v;
        } else {
          outH[idx] = (_Float16)v;
        }
      }
    }
  }
}

// ----------------------------------------------------------- RoPE + scatter
__global__ __launch_bounds__(256)
void rope_scatter_kernel(const _Float16* __restrict__ qp,
                         const _Float16* __restrict__ kp,
                         const _Float16* __restrict__ vp,
                         const int* __restrict__ off,
                         const int* __restrict__ eoff,
                         _Float16* __restrict__ q16,
                         _Float16* __restrict__ ck16,
                         _Float16* __restrict__ cvT) {
  int bt = blockIdx.x;  // over B*T
  int b = bt / kT, t = bt % kT;
  int p    = off[b] + t;
  int slot = (eoff[b] + t) % kCAP;
  for (int u = threadIdx.x; u < kH * (kDH / 2); u += 256) {
    int h = u >> 5;        // kDH/2 = 32
    int i = u & 31;
    float inv = __powf(10000.f, -(float)i / 32.f);
    float ang = (float)p * inv;
    float sn, cs;
    __sincosf(ang, &sn, &cs);
    size_t src = (size_t)bt * kD + h * kDH + 2 * i;
    float q1 = (float)qp[src], q2 = (float)qp[src + 1];
    float k1 = (float)kp[src], k2 = (float)kp[src + 1];
    float v1 = (float)vp[src], v2 = (float)vp[src + 1];
    size_t bh = (size_t)(b * kH + h);
    size_t qd = (bh * kT + t) * kDH + 2 * i;
    q16[qd]     = (_Float16)(q1 * cs - q2 * sn);
    q16[qd + 1] = (_Float16)(q2 * cs + q1 * sn);
    size_t kd = (bh * kCAP + slot) * kDH + 2 * i;
    ck16[kd]     = (_Float16)(k1 * cs - k2 * sn);
    ck16[kd + 1] = (_Float16)(k2 * cs + k1 * sn);
    size_t vd = (bh * kDH + 2 * i) * kCAP + slot;
    cvT[vd]        = (_Float16)v1;
    cvT[vd + kCAP] = (_Float16)v2;
  }
}

// ---------------------------------------------------------------- attention
// One wave handles a 16-row Q tile; flash-style online softmax over the
// ring cache in s-steps of 32. 4 score WMMAs + 4 PV WMMAs per step.
constexpr int kPS = 40;  // LDS row stride (halves) for the P tile
__global__ __launch_bounds__(128)
void attn_kernel(const _Float16* __restrict__ q16,
                 const _Float16* __restrict__ ck,
                 const _Float16* __restrict__ cvT,
                 const int* __restrict__ pos, const int* __restrict__ off,
                 _Float16* __restrict__ attn_out) {
  int lane = threadIdx.x & 31;
  int wave = threadIdx.x >> 5;
  int b = blockIdx.z, h = blockIdx.y;
  int t0 = (blockIdx.x * 4 + wave) * 16;
  size_t bh = (size_t)(b * kH + h);
  const _Float16* Q  = q16 + bh * kT * kDH;    // [T][DH]
  const _Float16* Kc = ck  + bh * kCAP * kDH;  // [CAP][DH]
  const _Float16* Vt = cvT + bh * kDH * kCAP;  // [DH][CAP]
  const int* posb = pos + b * kCAP;
  int offb = off[b];

  __shared__ _Float16 Plds[4][16 * kPS];
  _Float16* P = Plds[wave];

  v16h a0 = load_a_frag(Q, kDH, t0, 0,  lane);
  v16h a1 = load_a_frag(Q, kDH, t0, 32, lane);

  v8f o[4] = {};
  float mrun[8], lrun[8];
#pragma unroll
  for (int r = 0; r < 8; ++r) { mrun[r] = -1e30f; lrun[r] = 0.f; }

  int n = lane & 15, hi = lane >> 4;
  const float scale = 0.125f;  // 1/sqrt(64)

  for (int s0 = 0; s0 < kCAP; s0 += 32) {
    // ---- scores S[16][32] = Q @ K^T (two 16x16 C tiles)
    v8f sc0 = {}, sc1 = {};
    sc0 = wmma_f16(a0, load_b_frag(Kc, kDH, s0,      0,  lane), sc0);
    sc0 = wmma_f16(a1, load_b_frag(Kc, kDH, s0,      32, lane), sc0);
    sc1 = wmma_f16(a0, load_b_frag(Kc, kDH, s0 + 16, 0,  lane), sc1);
    sc1 = wmma_f16(a1, load_b_frag(Kc, kDH, s0 + 16, 32, lane), sc1);

    int ps0 = posb[s0 + n];
    int ps1 = posb[s0 + 16 + n];

    float p0[8], p1[8];
#pragma unroll
    for (int r = 0; r < 8; ++r) {
      int t  = t0 + r + 8 * hi;
      int pq = offb + t;
      int d0 = pq - ps0, d1 = pq - ps1;
      bool ok0 = (ps0 >= 0) && (d0 >= 0) && (d0 < kCTX);
      bool ok1 = (ps1 >= 0) && (d1 >= 0) && (d1 < kCTX);
      float v0 = ok0 ? sc0[r] * scale : -1e30f;
      float v1 = ok1 ? sc1[r] * scale : -1e30f;
      float mx = fmaxf(v0, v1);
#pragma unroll
      for (int msk = 1; msk < 16; msk <<= 1) mx = fmaxf(mx, __shfl_xor(mx, msk, 32));
      float mnew = fmaxf(mrun[r], mx);
      float e0 = ok0 ? __expf(v0 - mnew) : 0.f;
      float e1 = ok1 ? __expf(v1 - mnew) : 0.f;
      float sm = e0 + e1;
#pragma unroll
      for (int msk = 1; msk < 16; msk <<= 1) sm += __shfl_xor(sm, msk, 32);
      float f = __expf(mrun[r] - mnew);
      lrun[r] = lrun[r] * f + sm;
      mrun[r] = mnew;
#pragma unroll
      for (int j = 0; j < 4; ++j) o[j][r] *= f;
      p0[r] = e0;
      p1[r] = e1;
    }

    // ---- stage P (f16) through LDS to re-shape C-layout -> A-layout
    __syncthreads();
#pragma unroll
    for (int r = 0; r < 8; ++r) {
      int m = r + 8 * hi;
      P[m * kPS + n]      = (_Float16)p0[r];
      P[m * kPS + 16 + n] = (_Float16)p1[r];
    }
    __syncthreads();
    v16h ap = load_a_frag(P, kPS, 0, 0, lane);

    // ---- O += P[16x32] @ V[32x64]
#pragma unroll
    for (int j = 0; j < 4; ++j) {
      v16h bv = load_b_frag(Vt, kCAP, j * 16, s0, lane);
      o[j] = wmma_f16(ap, bv, o[j]);
    }
  }

  // ---- normalize, write in [B,T,D] layout (folds the transpose+reshape)
#pragma unroll
  for (int j = 0; j < 4; ++j) {
#pragma unroll
    for (int r = 0; r < 8; ++r) {
      int t = t0 + r + 8 * hi;
      int d = j * 16 + n;
      float val = (lrun[r] > 0.f) ? o[j][r] / lrun[r] : 0.f;
      attn_out[((size_t)b * kT + t) * kD + h * kDH + d] = (_Float16)val;
    }
  }
}

// --------------------------------------------------------------------- host
extern "C" void kernel_launch(void* const* d_in, const int* in_sizes, int n_in,
                              void* d_out, int out_size, void* d_ws,
                              size_t ws_size, hipStream_t stream) {
  (void)in_sizes; (void)n_in; (void)out_size; (void)ws_size;
  const float* x      = (const float*)d_in[0];
  const int*   offset = (const int*)d_in[1];
  const int*   eoff   = (const int*)d_in[2];
  const float* cache  = (const float*)d_in[3];
  const float* Wq     = (const float*)d_in[4];
  const float* Wk     = (const float*)d_in[5];
  const float* Wv     = (const float*)d_in[6];
  const float* Wo     = (const float*)d_in[7];
  const float* W1     = (const float*)d_in[8];
  const float* W2     = (const float*)d_in[9];
  const float* ln1_g  = (const float*)d_in[10];
  const float* ln1_b  = (const float*)d_in[11];
  const float* ln2_g  = (const float*)d_in[12];
  const float* ln2_b  = (const float*)d_in[13];
  float* out = (float*)d_out;

  char* wp = (char*)d_ws;
  auto carve = [&](size_t bytes) -> char* {
    char* r = wp;
    wp += (bytes + 255) & ~(size_t)255;
    return r;
  };
  const size_t szBT_D_h = (size_t)kBT * kD * sizeof(_Float16);
  _Float16* h16   = (_Float16*)carve(szBT_D_h);
  _Float16* wq16  = (_Float16*)carve((size_t)kD * kD * 2);
  _Float16* wk16  = (_Float16*)carve((size_t)kD * kD * 2);
  _Float16* wv16  = (_Float16*)carve((size_t)kD * kD * 2);
  _Float16* wo16  = (_Float16*)carve((size_t)kD * kD * 2);
  _Float16* w1_16 = (_Float16*)carve((size_t)kFF * kD * 2);
  _Float16* w2_16 = (_Float16*)carve((size_t)kD * kFF * 2);
  _Float16* qpre  = (_Float16*)carve(szBT_D_h);
  _Float16* kpre  = (_Float16*)carve(szBT_D_h);
  _Float16* vpre  = (_Float16*)carve(szBT_D_h);
  _Float16* q16   = (_Float16*)carve(szBT_D_h);
  const size_t szKV = (size_t)kB * kH * kCAP * kDH * 2;
  _Float16* ck16  = (_Float16*)carve(szKV);
  _Float16* cvT   = (_Float16*)carve(szKV);
  int*      posb  = (int*)carve((size_t)kB * kCAP * 4);
  _Float16* att   = (_Float16*)carve(szBT_D_h);
  float*    x1    = (float*)carve((size_t)kBT * kD * 4);
  _Float16* h2    = (_Float16*)carve(szBT_D_h);
  _Float16* u16   = (_Float16*)carve((size_t)kBT * kFF * 2);

  // 1) convert weights + cache, build position table
  auto cvt = [&](const float* src, _Float16* dst, int n) {
    f32_to_f16_kernel<<<(n + 255) / 256, 256, 0, stream>>>(src, dst, n);
  };
  cvt(Wq, wq16, kD * kD);
  cvt(Wk, wk16, kD * kD);
  cvt(Wv, wv16, kD * kD);
  cvt(Wo, wo16, kD * kD);
  cvt(W1, w1_16, kFF * kD);
  cvt(W2, w2_16, kD * kFF);
  {
    int total = kB * kH * kCAP * kDH;
    cache_cvt_kernel<<<(total + 255) / 256, 256, 0, stream>>>(cache, ck16, cvT, total);
  }
  pos_kernel<<<(kB * kCAP) / 256, 256, 0, stream>>>(eoff, posb);

  // 2) LN1 + QKV projections
  ln_kernel<<<kBT, 256, 0, stream>>>(x, ln1_g, ln1_b, h16, kD);
  dim3 gD(kD / 256, kBT / 32);
  gemm_kernel<0><<<gD, 128, 0, stream>>>(h16, wq16, kBT, kD, kD, qpre, nullptr, nullptr);
  gemm_kernel<0><<<gD, 128, 0, stream>>>(h16, wk16, kBT, kD, kD, kpre, nullptr, nullptr);
  gemm_kernel<0><<<gD, 128, 0, stream>>>(h16, wv16, kBT, kD, kD, vpre, nullptr, nullptr);

  // 3) RoPE + ring-buffer scatter into f16 K cache / transposed V cache
  rope_scatter_kernel<<<kBT, 256, 0, stream>>>(qpre, kpre, vpre, offset, eoff,
                                               q16, ck16, cvT);

  // 4) streaming attention
  attn_kernel<<<dim3(kT / 64, kH, kB), 128, 0, stream>>>(q16, ck16, cvT, posb,
                                                         offset, att);

  // 5) output projection + residual
  gemm_kernel<2><<<gD, 128, 0, stream>>>(att, wo16, kBT, kD, kD, nullptr, x1, x);

  // 6) LN2 + FFN (GELU) + residual into d_out
  ln_kernel<<<kBT, 256, 0, stream>>>(x1, ln2_g, ln2_b, h2, kD);
  gemm_kernel<1><<<dim3(kFF / 256, kBT / 32), 128, 0, stream>>>(
      h2, w1_16, kBT, kFF, kD, u16, nullptr, nullptr);
  gemm_kernel<2><<<gD, 128, 0, stream>>>(u16, w2_16, kBT, kD, kFF, nullptr, out, x1);
}
